// DKO_42185168781536
// MI455X (gfx1250) — matrix operand order repaired and checked by reading, not compiled, not run on hardware
//
#include <hip/hip_runtime.h>

typedef __bf16 bf16;
typedef __attribute__((ext_vector_type(16))) __bf16 v16bf;
typedef __attribute__((ext_vector_type(8)))  __bf16 v8bf;
typedef __attribute__((ext_vector_type(8)))  float  v8f;

union V16 { v16bf v; v8bf h[2]; };

static __device__ inline v8f wmma_bf16(v16bf a, v16bf b, v8f c) {
  // D = A(16x32 bf16) x B(32x16 bf16) + C(16x16 f32)
  return __builtin_amdgcn_wmma_f32_16x16x32_bf16(
      /*neg_a=*/false, a, /*neg_b=*/false, b,
      /*c_mod=*/(short)0, c, /*reuse_a=*/false, /*reuse_b=*/false);
}

// ---------------------------------------------------------------------------
// Kernel 0: bf16-pack A and Bm (zero-padded to K=32); zero the loss cell.
// ---------------------------------------------------------------------------
__global__ void pack_consts(const float* __restrict__ A, const float* __restrict__ Bm,
                            bf16* __restrict__ Abf, bf16* __restrict__ BmT,
                            float* __restrict__ out) {
  int i = blockIdx.x * blockDim.x + threadIdx.x;   // 0..65535
  if (i < 256 * 256) Abf[i] = (bf16)A[i];          // [n][k] row-major
  if (i < 256 * 32) {
    int n = i >> 5, k = i & 31;
    BmT[i] = (bf16)((k < 16) ? Bm[n * 16 + k] : 0.0f);   // [n][k], K padded to 32
  }
  if (i == 0) out[0] = 0.0f;
}

// ---------------------------------------------------------------------------
// Kernel 1: z0 = relu(x[:,7,:] @ W0^T + b0) @ W1^T + b1   -> ws (f32 [256][256])
// ---------------------------------------------------------------------------
__global__ __launch_bounds__(256) void mlp_z0(const float* __restrict__ x,
                                              const float* __restrict__ W0,
                                              const float* __restrict__ b0v,
                                              const float* __restrict__ W1,
                                              const float* __restrict__ b1v,
                                              float* __restrict__ z0) {
  __shared__ float sx[32];
  __shared__ float sh[512];
  const int b = blockIdx.x, tid = threadIdx.x;
  if (tid < 32) sx[tid] = x[((size_t)b * 1032 + 7) * 32 + tid];
  __syncthreads();
  for (int j = tid; j < 512; j += 256) {
    float acc = b0v[j];
    #pragma unroll
    for (int s = 0; s < 32; ++s) acc += sx[s] * W0[j * 32 + s];
    sh[j] = fmaxf(acc, 0.0f);
  }
  __syncthreads();
  float acc = b1v[tid];
  for (int j = 0; j < 512; ++j) acc += sh[j] * W1[tid * 512 + j];
  z0[(size_t)b * 256 + tid] = acc;
}

// ---------------------------------------------------------------------------
// Kernel 2: 1024-step scan. 16 WGs x 16 batch rows, 320 threads = 10 waves.
// Waves 0-7: recurrence GEMM (A slices register-resident, 2 tiles each).
// Waves 8-9: y = z' Cm^T, y stores and loss -- fully off the barrier path.
// LDS: z tile (8KB) + Cm operands (16KB).
// ---------------------------------------------------------------------------
__global__ __launch_bounds__(320, 1) void ssm_scan(
    const float* __restrict__ x, const float* __restrict__ u,
    const bf16* __restrict__ Abf, const float* __restrict__ z0,
    const bf16* __restrict__ BmT, const float* __restrict__ Cm,
    float* __restrict__ out) {
  __shared__ __attribute__((aligned(64))) bf16 sZ[16 * 256];   // z[m][k] bf16, 8KB
  __shared__ __attribute__((aligned(64))) bf16 sC[32 * 256];   // Cm[s][n] bf16, 16KB

  const int tid  = threadIdx.x;
  const int wid  = tid >> 5;
  const int lane = tid & 31;
  const int m    = lane & 15;   // row (A-op) / column (B-op) within tile
  const int hi   = lane >> 4;   // lane half select
  const int b0r  = blockIdx.x * 16;
  const bool isZ = (wid < 8);

  // ---- stage z0 tile and Cm to LDS ----
  for (int i = tid; i < 16 * 256; i += 320) sZ[i] = (bf16)z0[(size_t)b0r * 256 + i];
  for (int i = tid; i < 32 * 256; i += 320) sC[i] = (bf16)Cm[i];

  const bf16* zRow = sZ + m * 256;                      // A-op row (z)

  // ---- z-wave loop-invariant B-operands -> registers ----
  const int nt0 = (wid & 7) * 2;                        // z-wave's 2 N-tiles
  v16bf bA[2][8];                                       // A^T slices, 128 VGPRs
  #pragma unroll
  for (int tt = 0; tt < 2; ++tt) {
    const bf16* base = Abf + ((size_t)((nt0 + tt) * 16 + m)) * 256 + hi * 16;
    #pragma unroll
    for (int kt = 0; kt < 8; ++kt)
      bA[tt][kt] = *(const v16bf*)(base + kt * 32);
  }
  V16 bu0, bu1;
  bu0.h[0] = *(const v8bf*)(BmT + ((nt0 + 0) * 16 + m) * 32 + hi * 8);
  bu0.h[1] = *(const v8bf*)(BmT + ((nt0 + 0) * 16 + m) * 32 + 16 + hi * 8);
  bu1.h[0] = *(const v8bf*)(BmT + ((nt0 + 1) * 16 + m) * 32 + hi * 8);
  bu1.h[1] = *(const v8bf*)(BmT + ((nt0 + 1) * 16 + m) * 32 + 16 + hi * 8);
  const v16bf bu0v = bu0.v;
  const v16bf bu1v = bu1.v;

  const bf16* cRow = sC + ((wid & 1) * 16 + m) * 256;   // y B-op row (waves 8,9)
  const float* uLane = u + ((size_t)(b0r + m) * 1031 + 7) * 16 + hi * 8;

  const bf16 BZ = (bf16)0.0f;
  const v8bf ZH = {BZ, BZ, BZ, BZ, BZ, BZ, BZ, BZ};
  float lossAcc = 0.0f;

  __syncthreads();

  for (int t = 0; t < 1024; ++t) {
    if (isZ) {
      // -- u_t A-operand: 8 real K values + 8 zero pad (shared by both tiles) --
      const float4* up = (const float4*)(uLane + (size_t)t * 16);
      float4 f0 = up[0], f1 = up[1];
      V16 au;
      au.h[0] = (v8bf){(bf16)f0.x, (bf16)f0.y, (bf16)f0.z, (bf16)f0.w,
                       (bf16)f1.x, (bf16)f1.y, (bf16)f1.z, (bf16)f1.w};
      au.h[1] = ZH;
      if (t + 1 < 1024) __builtin_prefetch(uLane + (size_t)(t + 1) * 16, 0, 1);

      // -- z' = z A^T + u Bm^T for this wave's two 16x16 tiles --
      v8f acc0 = {0.f, 0.f, 0.f, 0.f, 0.f, 0.f, 0.f, 0.f};
      v8f acc1 = {0.f, 0.f, 0.f, 0.f, 0.f, 0.f, 0.f, 0.f};
      #pragma unroll
      for (int kt = 0; kt < 8; ++kt) {
        const int k0 = kt * 32;
        V16 a;
        a.h[0] = *(const v8bf*)(zRow + k0 + hi * 16);       // K=k0+(hi?16..23:0..7)
        a.h[1] = *(const v8bf*)(zRow + k0 + hi * 16 + 8);   // K=k0+(hi?24..31:8..15)
        acc0 = wmma_bf16(a.v, bA[0][kt], acc0);
        acc1 = wmma_bf16(a.v, bA[1][kt], acc1);
      }
      acc0 = wmma_bf16(au.v, bu0v, acc0);
      acc1 = wmma_bf16(au.v, bu1v, acc1);

      __syncthreads();   // z_t reads done (z-waves t, y-waves t-1)

      // -- write z_{t+1} back to LDS (bf16 carry) --
      #pragma unroll
      for (int r = 0; r < 8; ++r) {
        const int row = (r + hi * 8) * 256;
        sZ[row + (nt0 + 0) * 16 + m] = (bf16)acc0[r];
        sZ[row + (nt0 + 1) * 16 + m] = (bf16)acc1[r];
      }
      __syncthreads();   // z_{t+1} visible
    } else {
      __syncthreads();   // match barrier 1
      __syncthreads();   // match barrier 2: z_{t+1} now visible

      // -- y_t = z_{t+1} Cm^T, one 16x16 S-tile per y-wave; overlaps next z step --
      v8f yac = {0.f, 0.f, 0.f, 0.f, 0.f, 0.f, 0.f, 0.f};
      #pragma unroll
      for (int kt = 0; kt < 8; ++kt) {
        const int k0 = kt * 32;
        V16 a;
        a.h[0] = *(const v8bf*)(zRow + k0 + hi * 16);
        a.h[1] = *(const v8bf*)(zRow + k0 + hi * 16 + 8);
        v16bf cc = *(const v16bf*)(cRow + k0 + hi * 16);
        yac = wmma_bf16(a.v, cc, yac);
      }
      const int s = (wid & 1) * 16 + m;
      if (t + 1 < 1024)
        __builtin_prefetch(&x[((size_t)(b0r + hi * 8) * 1032 + 9 + t) * 32 + s], 0, 1);
      #pragma unroll
      for (int r = 0; r < 8; ++r) {
        const int b = b0r + r + hi * 8;
        const float yv = yac[r];
        out[1 + (((size_t)t * 256 + b) * 32 + s)] = yv;
        const float xv = x[((size_t)b * 1032 + 8 + t) * 32 + s];
        const float d = yv - xv;
        lossAcc += d * d;
      }
    }
  }

  // -- loss: wave reduce + one atomic per y-wave --
  float v = lossAcc;
  #pragma unroll
  for (int off = 16; off > 0; off >>= 1) v += __shfl_down(v, off, 32);
  if (wid >= 8 && lane == 0)
    atomicAdd(out, v * (1.0f / (1024.0f * 256.0f * 32.0f)));
}

// ---------------------------------------------------------------------------
extern "C" void kernel_launch(void* const* d_in, const int* in_sizes, int n_in,
                              void* d_out, int out_size, void* d_ws, size_t ws_size,
                              hipStream_t stream) {
  const float* x   = (const float*)d_in[0];
  const float* u   = (const float*)d_in[1];
  const float* A   = (const float*)d_in[2];
  const float* Bm  = (const float*)d_in[3];
  const float* Cm  = (const float*)d_in[4];
  const float* W0  = (const float*)d_in[5];
  const float* b0v = (const float*)d_in[6];
  const float* W1  = (const float*)d_in[7];
  const float* b1v = (const float*)d_in[8];
  float* out = (float*)d_out;

  char* ws = (char*)d_ws;
  float* z0  = (float*)ws;                        // 256*256 f32  = 262144 B
  bf16*  Abf = (bf16*)(ws + 262144);              // 256*256 bf16 = 131072 B
  bf16*  BmT = (bf16*)(ws + 262144 + 131072);     // 256*32  bf16 =  16384 B

  hipLaunchKernelGGL(pack_consts, dim3(256), dim3(256), 0, stream, A, Bm, Abf, BmT, out);
  hipLaunchKernelGGL(mlp_z0,      dim3(256), dim3(256), 0, stream, x, W0, b0v, W1, b1v, z0);
  hipLaunchKernelGGL(ssm_scan,    dim3(16),  dim3(320), 0, stream, x, u, Abf, z0, BmT, Cm, out);
}
